// SimpleEdgeBlock_40724879901211
// MI455X (gfx1250) — matrix operand order repaired and checked by reading, not compiled, not run on hardware
//
#include <hip/hip_runtime.h>

// ---------------------------------------------------------------------------
// FullyConnectedTensorProduct(64x0e+64x1o)^2 -> 64x0e+64x1o, summed over 2 TPs
// A = per-edge outer products generated on the fly in packed bf16 registers,
// B = pre-transposed bf16 weights (double-buffered from L2),
// f32 accumulation via v_wmma_f32_16x16x32_bf16.
// ---------------------------------------------------------------------------

typedef __attribute__((ext_vector_type(16))) __bf16 v16bf;
typedef __attribute__((ext_vector_type(2)))  __bf16 bf16x2;
typedef __attribute__((ext_vector_type(8)))  float  v8f;

#define DIMTOT   256           // 64 scalars + 64*3 vectors
#define TILE_E   16            // edges per workgroup (WMMA M)
#define XSTR     260           // padded LDS row stride in floats (conflict-free)
#define WPATH    262144        // 64*64*64 elements per weight tensor

// pack two floats into one VGPR of 2x bf16 (v_cvt_pk_bf16_f32)
static __device__ __forceinline__ unsigned pack2(float a, float b) {
    bf16x2 p = {(__bf16)a, (__bf16)b};
    return __builtin_bit_cast(unsigned, p);
}
// guaranteed packed-bf16 VALU ops (VOP3P): 2 A-elements per instruction
static __device__ __forceinline__ unsigned pkmul(unsigned a, unsigned b) {
    unsigned d;
    asm("v_pk_mul_bf16 %0, %1, %2" : "=v"(d) : "v"(a), "v"(b));
    return d;
}
static __device__ __forceinline__ unsigned pkfma(unsigned a, unsigned b, unsigned c) {
    unsigned d;
    asm("v_pk_fma_bf16 %0, %1, %2, %3" : "=v"(d) : "v"(a), "v"(b), "v"(c));
    return d;
}
union Afrag { unsigned u[8]; v16bf v; };

// ---------------------------------------------------------------------------
// Prep: W[u][v][w] (f32) -> Wt[path][w][u*64+v] (bf16): per-lane B slice is
// 16 contiguous bf16 (32B -> two global_load_b128).
// ---------------------------------------------------------------------------
__global__ __launch_bounds__(256) void prep_weights(
    const float* __restrict__ w0, const float* __restrict__ w1,
    const float* __restrict__ w2, const float* __restrict__ w3,
    const float* __restrict__ w4, const float* __restrict__ w5,
    const float* __restrict__ w6, const float* __restrict__ w7,
    __bf16* __restrict__ wt)
{
    size_t g = (size_t)blockIdx.x * blockDim.x + threadIdx.x;
    if (g >= (size_t)8 * WPATH) return;
    int p   = (int)(g >> 18);
    int rem = (int)(g & (WPATH - 1));
    const float* W;
    switch (p) {
        case 0: W = w0; break; case 1: W = w1; break;
        case 2: W = w2; break; case 3: W = w3; break;
        case 4: W = w4; break; case 5: W = w5; break;
        case 6: W = w6; break; default: W = w7; break;
    }
    int w  = rem >> 12;        // 0..63
    int uv = rem & 4095;       // u*64+v
    wt[g] = (__bf16)W[(size_t)uv * 64 + w];   // source flat = uv*64 + w
}

// ---------------------------------------------------------------------------
// Main kernel: one 16-edge tile per workgroup, 4 waves x one 16-wide n-tile.
// ---------------------------------------------------------------------------
__global__ __launch_bounds__(128) void fctp_kernel(
    const float* __restrict__ e1x, const float* __restrict__ e1y,
    const float* __restrict__ e2x, const float* __restrict__ e2y,
    const __bf16* __restrict__ wt, float* __restrict__ out, int E)
{
    extern __shared__ float lds[];   // 4 tensors x [16][XSTR] floats

    const int tid  = threadIdx.x;
    const int wid  = tid >> 5;       // 0..3 -> n-tile
    const int lane = tid & 31;
    const int er   = lane & 15;      // A row (edge) == B column (w) lane index
    const int half = lane >> 4;
    const int n0   = wid * 16;
    const long tileBase = (long)blockIdx.x * TILE_E;

    // ---- stage x/y tiles into padded LDS ----
    const float* srcs[4] = {e1x, e1y, e2x, e2y};
    #pragma unroll
    for (int t = 0; t < 4; ++t) {
        float* dst = lds + t * (TILE_E * XSTR);
        for (int idx = tid; idx < TILE_E * 64; idx += 128) {   // float4 granules
            int r = idx >> 6, q = idx & 63;
            long eg = tileBase + r; if (eg >= E) eg = E - 1;   // clamp (harmless dup)
            const float4 val = ((const float4*)(srcs[t] + eg * DIMTOT))[q];
            ((float4*)(dst + r * XSTR))[q] = val;
        }
    }
    __syncthreads();

    // path normalization constants (folded into A-side x scalars).
    // NOTE: C110 == C011 == C101, so one scaled x1 triple serves paths 110 & 101.
    constexpr float C000 = 0.011048543456039806f;              // 1/sqrt(2*64*64)
    constexpr float C3   = C000 * 0.57735026918962576f;        // * 1/sqrt(3)

    v8f acc0  = {};                              // out0 (w = n0..n0+15)
    v8f acc1a = {}, acc1b = {}, acc1c = {};      // out1 components i=0,1,2

    #pragma unroll 1
    for (int tp = 0; tp < 2; ++tp) {
        const float* X = lds + (tp ? 2 : 0) * (TILE_E * XSTR);
        const float* Y = lds + (tp ? 3 : 1) * (TILE_E * XSTR);
        const size_t wb = (size_t)(tp * 4) * WPATH + (size_t)(n0 + er) * 4096 + 16 * half;

        #pragma unroll 1
        for (int s = 0; s < 2; ++s) {
            const int vb = s * 32;
            const __bf16* pW000 = wt + wb + vb;
            const __bf16* pW110 = wt + wb + vb + WPATH;
            const __bf16* pW011 = wt + wb + vb + 2 * (size_t)WPATH;
            const __bf16* pW101 = wt + wb + vb + 3 * (size_t)WPATH;

            // hoisted y rows, pre-converted to packed bf16 pairs.
            // A layout: element j <-> K = j + 16*half; pair q = elems 2q,2q+1
            unsigned y0p[8], y1p0[8], y1p1[8], y1p2[8];
            #pragma unroll
            for (int q = 0; q < 8; ++q) {
                const int j = 2 * q;
                const int v = vb + j + 16 * half;
                const int yb = er * XSTR;
                y0p[q]  = pack2(Y[yb + v],              Y[yb + v + 1]);
                y1p0[q] = pack2(Y[yb + 64 + v * 3 + 0], Y[yb + 64 + v * 3 + 3]);
                y1p1[q] = pack2(Y[yb + 64 + v * 3 + 1], Y[yb + 64 + v * 3 + 4]);
                y1p2[q] = pack2(Y[yb + 64 + v * 3 + 2], Y[yb + 64 + v * 3 + 5]);
            }

            // ---- B double buffer: prime u=0 ----
            v16bf B000c = *(const v16bf*)(pW000);
            v16bf B110c = *(const v16bf*)(pW110);
            v16bf B011c = *(const v16bf*)(pW011);
            v16bf B101c = *(const v16bf*)(pW101);

            #pragma unroll 2
            for (int u = 0; u < 64; ++u) {
                // prefetch next u (wraps at the end; value unused then)
                const size_t kon = (size_t)(((u + 1) & 63) * 64);
                const v16bf B000n = *(const v16bf*)(pW000 + kon);
                const v16bf B110n = *(const v16bf*)(pW110 + kon);
                const v16bf B011n = *(const v16bf*)(pW011 + kon);
                const v16bf B101n = *(const v16bf*)(pW101 + kon);

                const float x0u  = X[er * XSTR + u];
                const float x1u0 = X[er * XSTR + 64 + u * 3 + 0];
                const float x1u1 = X[er * XSTR + 64 + u * 3 + 1];
                const float x1u2 = X[er * XSTR + 64 + u * 3 + 2];

                // broadcast-packed, path-constant-scaled x scalars (bf16x2)
                const unsigned xs000 = pack2(x0u * C000, x0u * C000);
                const unsigned xs011 = pack2(x0u * C3,   x0u * C3);
                const unsigned t0p   = pack2(x1u0 * C3,  x1u0 * C3);  // paths 110 & 101
                const unsigned t1p   = pack2(x1u1 * C3,  x1u1 * C3);
                const unsigned t2p   = pack2(x1u2 * C3,  x1u2 * C3);

                Afrag A;
                // path (0e,0e)->0e : z = x0[u]*y0[v]
                #pragma unroll
                for (int q = 0; q < 8; ++q) A.u[q] = pkmul(xs000, y0p[q]);
                acc0 = __builtin_amdgcn_wmma_f32_16x16x32_bf16(false, A.v, false, B000c,
                                                               (short)0, acc0, false, false);
                // path (1o,1o)->0e : z = dot3(x1[u], y1[v])
                #pragma unroll
                for (int q = 0; q < 8; ++q)
                    A.u[q] = pkfma(t2p, y1p2[q], pkfma(t1p, y1p1[q], pkmul(t0p, y1p0[q])));
                acc0 = __builtin_amdgcn_wmma_f32_16x16x32_bf16(false, A.v, false, B110c,
                                                               (short)0, acc0, false, false);
                // path (0e,1o)->1o : z_i = x0[u]*y1[v,i]   (B011 shared across i)
                #pragma unroll
                for (int q = 0; q < 8; ++q) A.u[q] = pkmul(xs011, y1p0[q]);
                acc1a = __builtin_amdgcn_wmma_f32_16x16x32_bf16(false, A.v, false, B011c,
                                                                (short)0, acc1a, false, false);
                #pragma unroll
                for (int q = 0; q < 8; ++q) A.u[q] = pkmul(xs011, y1p1[q]);
                acc1b = __builtin_amdgcn_wmma_f32_16x16x32_bf16(false, A.v, false, B011c,
                                                                (short)0, acc1b, false, false);
                #pragma unroll
                for (int q = 0; q < 8; ++q) A.u[q] = pkmul(xs011, y1p2[q]);
                acc1c = __builtin_amdgcn_wmma_f32_16x16x32_bf16(false, A.v, false, B011c,
                                                                (short)0, acc1c, false, false);
                // path (1o,0e)->1o : z_i = x1[u,i]*y0[v]   (B101 shared across i)
                #pragma unroll
                for (int q = 0; q < 8; ++q) A.u[q] = pkmul(t0p, y0p[q]);
                acc1a = __builtin_amdgcn_wmma_f32_16x16x32_bf16(false, A.v, false, B101c,
                                                                (short)0, acc1a, false, false);
                #pragma unroll
                for (int q = 0; q < 8; ++q) A.u[q] = pkmul(t1p, y0p[q]);
                acc1b = __builtin_amdgcn_wmma_f32_16x16x32_bf16(false, A.v, false, B101c,
                                                                (short)0, acc1b, false, false);
                #pragma unroll
                for (int q = 0; q < 8; ++q) A.u[q] = pkmul(t2p, y0p[q]);
                acc1c = __builtin_amdgcn_wmma_f32_16x16x32_bf16(false, A.v, false, B101c,
                                                                (short)0, acc1c, false, false);

                // rotate double buffer
                B000c = B000n; B110c = B110n; B011c = B011n; B101c = B101n;
            }
        }
    }

    // ---- store D tiles: M = r + 8*half (edge), N = lane&15 (w within n-tile) ----
    struct F3 { float a, b, c; };
    #pragma unroll
    for (int r = 0; r < 8; ++r) {
        const long eg = tileBase + r + 8 * half;
        if (eg >= E) continue;
        float* o = out + eg * DIMTOT;
        o[n0 + er] = acc0[r];                                     // out0, contiguous
        F3 v3 = {acc1a[r], acc1b[r], acc1c[r]};
        *(F3*)(o + 64 + (size_t)(n0 + er) * 3) = v3;              // out1, b96 per lane
    }
}

// ---------------------------------------------------------------------------
extern "C" void kernel_launch(void* const* d_in, const int* in_sizes, int n_in,
                              void* d_out, int out_size, void* d_ws, size_t ws_size,
                              hipStream_t stream) {
    const float* e1x = (const float*)d_in[0];
    const float* e1y = (const float*)d_in[1];
    const float* e2x = (const float*)d_in[2];
    const float* e2y = (const float*)d_in[3];
    // weights in setup order: W1_000,W1_110,W1_011,W1_101,W2_000,W2_110,W2_011,W2_101
    __bf16* wt = (__bf16*)d_ws;   // needs 8*262144*2 = 4 MiB of workspace

    prep_weights<<<(8 * WPATH + 255) / 256, 256, 0, stream>>>(
        (const float*)d_in[4], (const float*)d_in[5], (const float*)d_in[6],
        (const float*)d_in[7], (const float*)d_in[8], (const float*)d_in[9],
        (const float*)d_in[10], (const float*)d_in[11], wt);

    const int E = in_sizes[0] / DIMTOT;            // 100000
    const int nTiles = (E + TILE_E - 1) / TILE_E;  // 6250
    const size_t shmem = (size_t)4 * TILE_E * XSTR * sizeof(float);  // 66,560 B

    (void)hipFuncSetAttribute((const void*)fctp_kernel,
                              hipFuncAttributeMaxDynamicSharedMemorySize, (int)shmem);
    fctp_kernel<<<nTiles, 128, shmem, stream>>>(e1x, e1y, e2x, e2y, wt,
                                                (float*)d_out, E);
}
